// Attention_17008070493052
// MI455X (gfx1250) — matrix operand order repaired and checked by reading, not compiled, not run on hardware
//
#include <hip/hip_runtime.h>

#define DM 768
#define SEQ 2048
#define BS 4
#define NH 12
#define DH 64

constexpr int    WN  = DM * DM;                    // 589824 elems per weight
constexpr size_t ACT = (size_t)BS * NH * SEQ * DH; // 6291456 elems per activation

typedef __attribute__((ext_vector_type(16))) __bf16 v16bf;
typedef __attribute__((ext_vector_type(8)))  float  v8f;

// ---------- fragment loaders ----------

// A-fragment (16x32 bf16), loading from f32 memory with inline conversion.
// Per-lane: elements 0..15 = K[koffA .. koffA+15] (contiguous)
__device__ inline v16bf fragA_f32(const float* p) {
  float4 a0 = *(const float4*)(p);
  float4 a1 = *(const float4*)(p + 4);
  float4 b0 = *(const float4*)(p + 8);
  float4 b1 = *(const float4*)(p + 12);
  v16bf r;
  r[0]  = (__bf16)a0.x; r[1]  = (__bf16)a0.y; r[2]  = (__bf16)a0.z; r[3]  = (__bf16)a0.w;
  r[4]  = (__bf16)a1.x; r[5]  = (__bf16)a1.y; r[6]  = (__bf16)a1.z; r[7]  = (__bf16)a1.w;
  r[8]  = (__bf16)b0.x; r[9]  = (__bf16)b0.y; r[10] = (__bf16)b0.z; r[11] = (__bf16)b0.w;
  r[12] = (__bf16)b1.x; r[13] = (__bf16)b1.y; r[14] = (__bf16)b1.z; r[15] = (__bf16)b1.w;
  return r;
}

// A-fragment from bf16 memory/LDS: two 16B chunks at p and p+8 elements.
__device__ inline v16bf fragA_bf16(const __bf16* p) {
  union U { uint4 u; __bf16 h[8]; } c0, c1;
  c0.u = *(const uint4*)(p);
  c1.u = *(const uint4*)(p + 8);
  v16bf r;
#pragma unroll
  for (int i = 0; i < 8; i++) { r[i] = c0.h[i]; r[8 + i] = c1.h[i]; }
  return r;
}

// B-fragment (32x16 bf16): per lane, 16 CONTIGUOUS K elements (K block 0-15 or 16-31).
__device__ inline v16bf fragB_bf16(const __bf16* p) {
  union U { uint4 u; __bf16 h[8]; } c0, c1;
  c0.u = *(const uint4*)(p);
  c1.u = *(const uint4*)(p + 8);
  v16bf r;
#pragma unroll
  for (int i = 0; i < 8; i++) { r[i] = c0.h[i]; r[8 + i] = c1.h[i]; }
  return r;
}

#define WMMA_BF16(A, B, C) \
  __builtin_amdgcn_wmma_f32_16x16x32_bf16(false, (A), false, (B), (short)0, (C), false, false)

// Async Global -> LDS copy of 16 bytes (CDNA5, ASYNCcnt-tracked, no VGPR round trip).
__device__ inline void async_copy_b128(unsigned ldsByteAddr, const void* gptr) {
  asm volatile("global_load_async_to_lds_b128 %0, %1, off"
               :
               : "v"(ldsByteAddr), "v"((unsigned long long)(uintptr_t)gptr)
               : "memory");
}

// ---------- kernel 1: weight f32 -> bf16 ----------
__global__ void cvt_w_kernel(const float* __restrict__ Wq, const float* __restrict__ Wk,
                             const float* __restrict__ Wv, __bf16* __restrict__ wbf) {
  int i = blockIdx.x * blockDim.x + threadIdx.x;
  if (i >= 3 * WN) return;
  int p = i / WN, r = i - p * WN;
  const float* src = (p == 0) ? Wq : (p == 1) ? Wk : Wv;
  wbf[i] = (__bf16)src[r];
}

// ---------- kernel 2: projection GEMM  C = X @ W^T + b ----------
// Wave computes a 16(M) x 64(N) tile, 4 accumulators, k-loop of 24 x WMMA(16x16x32).
// proj 0/1 (Q/K): output bf16 [b][h][s][d]; proj 2 (V): output bf16 [b][h][d][s] (transposed).
__global__ void proj_kernel(const float* __restrict__ q, const float* __restrict__ k,
                            const float* __restrict__ v,
                            const float* __restrict__ bq, const float* __restrict__ bk,
                            const float* __restrict__ bv,
                            const __bf16* __restrict__ wbf,
                            __bf16* __restrict__ Qp, __bf16* __restrict__ Kp,
                            __bf16* __restrict__ VpT) {
  const int lane  = threadIdx.x & 31;
  const int wid   = threadIdx.x >> 5;
  const int p     = blockIdx.z;
  const int l15   = lane & 15;
  const int koffA = (lane & 16) ? 16 : 0;
  const int koffB = (lane & 16) ? 16 : 0;
  const int rhalf = (lane >> 4) << 3;

  const float*  X    = (p == 0) ? q  : (p == 1) ? k  : v;
  const float*  bias = (p == 0) ? bq : (p == 1) ? bk : bv;
  const __bf16* W    = wbf + (size_t)p * WN;

  const int m0    = (blockIdx.x * 4 + wid) * 16;
  const int nBase = blockIdx.y * 64;

  v8f acc[4] = {};
  const float* arow = X + (size_t)(m0 + l15) * DM;

  for (int kk = 0; kk < DM; kk += 32) {
    v16bf af = fragA_f32(arow + kk + koffA);
#pragma unroll
    for (int t = 0; t < 4; t++) {
      const __bf16* bp = W + (size_t)(nBase + t * 16 + l15) * DM + kk + koffB;
      v16bf bf = fragB_bf16(bp);
      acc[t] = WMMA_BF16(af, bf, acc[t]);
    }
  }

#pragma unroll
  for (int t = 0; t < 4; t++) {
    const int n   = nBase + t * 16 + l15;
    const float bvv = bias[n];
    const int h = n >> 6, d = n & 63;
#pragma unroll
    for (int r = 0; r < 8; r++) {
      const int m = m0 + r + rhalf;
      const int b = m >> 11, s = m & (SEQ - 1);
      const float val = acc[t][r] + bvv;
      if (p == 2) {
        VpT[(((size_t)b * NH + h) * DH + d) * SEQ + s] = (__bf16)val;
      } else {
        __bf16* dst = (p == 0) ? Qp : Kp;
        dst[(((size_t)b * NH + h) * SEQ + s) * DH + d] = (__bf16)val;
      }
    }
  }
}

// ---------- kernel 3: flash attention with async-LDS staged K/V tiles ----------
// Block = 4 waves sharing one (b,h); each wave owns one 16-row q tile.
// Per 32-key step: K tile (32x64) and V^T tile (64x32) are DMA'd to LDS once per
// block with global_load_async_to_lds_b128 (double buffered), then 4 score WMMAs +
// online softmax + LDS P-transpose + 4 PV WMMAs.
__global__ void attn_kernel(const __bf16* __restrict__ Qp, const __bf16* __restrict__ Kp,
                            const __bf16* __restrict__ VpT, float* __restrict__ out) {
  __shared__ __bf16 KtS[2][32 * DH];   // [buf][j'][d]
  __shared__ __bf16 VtS[2][DH * 32];   // [buf][d][j']
  __shared__ __bf16 ldsP[4 * 16 * 32]; // per-wave P staging

  const int tid   = threadIdx.x;
  const int lane  = tid & 31;
  const int wid   = tid >> 5;
  const int l15   = lane & 15;
  const int koffA = (lane & 16) ? 16 : 0;
  const int koffB = (lane & 16) ? 16 : 0;
  const int rhalf = (lane >> 4) << 3;

  const int h  = blockIdx.y;
  const int b  = blockIdx.z;
  const int bh = b * NH + h;
  const int qs = (blockIdx.x * 4 + wid) * 16;

  const __bf16* Qb  = Qp  + (size_t)bh * SEQ * DH;
  const __bf16* Kb  = Kp  + (size_t)bh * SEQ * DH;
  const __bf16* Vtb = VpT + (size_t)bh * DH * SEQ;
  __bf16* myP = ldsP + wid * (16 * 32);

  // LDS byte offsets (low 32 bits of the generic pointer == LDS address)
  const unsigned kOff[2] = { (unsigned)(uintptr_t)&KtS[0][0], (unsigned)(uintptr_t)&KtS[1][0] };
  const unsigned vOff[2] = { (unsigned)(uintptr_t)&VtS[0][0], (unsigned)(uintptr_t)&VtS[1][0] };

  // cooperative async tile load: K tile is one contiguous 4KB block; V^T tile is
  // 64 rows of 64B (row stride SEQ elements).
  auto load_tile = [&](int buf, int j0) {
    const char* kg = (const char*)(Kb + (size_t)j0 * DH);
    async_copy_b128(kOff[buf] + (unsigned)tid * 16u, kg + (size_t)tid * 16);
    async_copy_b128(kOff[buf] + (unsigned)(tid + 128) * 16u, kg + (size_t)(tid + 128) * 16);
#pragma unroll
    for (int cc = 0; cc < 2; cc++) {
      int c   = tid + cc * 128;     // 256 chunks of 16B
      int row = c >> 2;             // d row 0..63
      int col = c & 3;              // 16B column within row
      const char* vg = (const char*)(Vtb + (size_t)row * SEQ + j0) + col * 16;
      async_copy_b128(vOff[buf] + (unsigned)c * 16u, vg);
    }
  };

  v16bf qf[2];
#pragma unroll
  for (int c = 0; c < 2; c++)
    qf[c] = fragA_bf16(Qb + (size_t)(qs + l15) * DH + c * 32 + koffA);

  float mrow[8], lrow[8];
  v8f o[4] = {};
#pragma unroll
  for (int r = 0; r < 8; r++) { mrow[r] = -3.0e38f; lrow[r] = 0.0f; }

  load_tile(0, 0);
  int cur = 0;

  for (int j = 0; j < SEQ; j += 32) {
    asm volatile("s_wait_asynccnt 0" ::: "memory");
    __syncthreads();                        // current tile ready in LDS for all waves
    if (j + 32 < SEQ) load_tile(cur ^ 1, j + 32);  // overlap next tile's DMA

    const __bf16* Ktile = &KtS[cur][0];
    const __bf16* Vtile = &VtS[cur][0];

    // scores: two 16x16 tiles over d = 0..63 (B-fragments from LDS)
    v8f s0 = {}, s1 = {};
#pragma unroll
    for (int c = 0; c < 2; c++) {
      v16bf kf0 = fragB_bf16(Ktile + (size_t)(l15) * DH + c * 32 + koffB);
      v16bf kf1 = fragB_bf16(Ktile + (size_t)(16 + l15) * DH + c * 32 + koffB);
      s0 = WMMA_BF16(qf[c], kf0, s0);
      s1 = WMMA_BF16(qf[c], kf1, s1);
    }

    // online softmax (rows striped: VGPR r <-> row r + rhalf, cols across 16 lanes)
#pragma unroll
    for (int r = 0; r < 8; r++) {
      float rm = fmaxf(s0[r], s1[r]);
#pragma unroll
      for (int off = 8; off >= 1; off >>= 1) rm = fmaxf(rm, __shfl_xor(rm, off, 32));
      const float mnew = fmaxf(mrow[r], rm);
      const float scl  = __expf(mrow[r] - mnew);
      const float p0   = __expf(s0[r] - mnew);
      const float p1   = __expf(s1[r] - mnew);
      float rs = p0 + p1;
#pragma unroll
      for (int off = 8; off >= 1; off >>= 1) rs += __shfl_xor(rs, off, 32);
      lrow[r] = lrow[r] * scl + rs;
      mrow[r] = mnew;
#pragma unroll
      for (int t = 0; t < 4; t++) o[t][r] *= scl;
      const int i = r + rhalf;
      myP[i * 32 + l15]      = (__bf16)p0;
      myP[i * 32 + 16 + l15] = (__bf16)p1;
    }

    asm volatile("s_wait_dscnt 0" ::: "memory");

    // P (16x32) from LDS in A layout, then P @ V over 4 d-tiles (V from LDS)
    v16bf pf = fragA_bf16(myP + l15 * 32 + koffA);
#pragma unroll
    for (int t = 0; t < 4; t++) {
      v16bf vf = fragB_bf16(Vtile + (size_t)(t * 16 + l15) * 32 + koffB);
      o[t] = WMMA_BF16(pf, vf, o[t]);
    }

    __syncthreads();                        // all waves done reading current buffers
    cur ^= 1;
  }

  // normalize + store f32 [b][s][h*64+d]
#pragma unroll
  for (int t = 0; t < 4; t++) {
#pragma unroll
    for (int r = 0; r < 8; r++) {
      const int row = qs + r + rhalf;
      out[((size_t)b * SEQ + row) * DM + h * DH + t * 16 + l15] = o[t][r] / lrow[r];
    }
  }
}

// ---------- host launcher ----------
extern "C" void kernel_launch(void* const* d_in, const int* in_sizes, int n_in,
                              void* d_out, int out_size, void* d_ws, size_t ws_size,
                              hipStream_t stream) {
  (void)in_sizes; (void)n_in; (void)out_size; (void)ws_size;
  const float* q  = (const float*)d_in[0];
  const float* k  = (const float*)d_in[1];
  const float* v  = (const float*)d_in[2];
  const float* Wq = (const float*)d_in[3];
  const float* bq = (const float*)d_in[4];
  const float* Wk = (const float*)d_in[5];
  const float* bk = (const float*)d_in[6];
  const float* Wv = (const float*)d_in[7];
  const float* bv = (const float*)d_in[8];
  float* out = (float*)d_out;

  char* ws = (char*)d_ws;
  __bf16* wbf = (__bf16*)ws;                              // 3*WN bf16
  __bf16* Qpb = (__bf16*)(ws + (size_t)3 * WN * 2);       // ACT bf16
  __bf16* Kpb = Qpb + ACT;                                // ACT bf16
  __bf16* VpT = Kpb + ACT;                                // ACT bf16

  // 1) convert weights to bf16
  {
    int n = 3 * WN;
    cvt_w_kernel<<<(n + 255) / 256, 256, 0, stream>>>(Wq, Wk, Wv, wbf);
  }
  // 2) QKV projections: grid = (M tiles/4, N groups of 64, proj)
  {
    dim3 grid((BS * SEQ) / 64, DM / 64, 3);
    proj_kernel<<<grid, 128, 0, stream>>>(q, k, v, bq, bk, bv, wbf, Qpb, Kpb, VpT);
  }
  // 3) flash attention: grid = (q tiles / 4 waves, heads, batch)
  {
    dim3 grid(SEQ / 64, NH, BS);
    attn_kernel<<<grid, 128, 0, stream>>>(Qpb, Kpb, VpT, out);
  }
}
